// DilatedCorrelation_2173253452155
// MI455X (gfx1250) — matrix operand order repaired and checked by reading, not compiled, not run on hardware
//
#include <hip/hip_runtime.h>

// ---- problem constants (from reference) ----
#define B_   8
#define C_   256
#define H_   128
#define W_   192
#define KC   32              // channels per LDS chunk (one WMMA K step)
#define NDY  7               // distinct row shifts
#define PW   200             // padded prev row width: x' in [-4, 196)
#define SA   40              // ushort stride per x entry (32 ch + 8 pad -> 80B, 16B aligned, conflict-free)
#define NWAVE 12             // one wave per 16-wide x tile (12*16 = 192)
#define TPB  (NWAVE * 32)
#define SCR  33              // f32 scratch row stride (odd -> conflict-free reads)

typedef __attribute__((ext_vector_type(16))) __bf16       v16bf;
typedef __attribute__((ext_vector_type(8)))  float        v8f;
typedef __attribute__((ext_vector_type(4)))  unsigned int v4u;

// offset tables (reference enumeration order: dy outer -4..4, dx inner)
__constant__ int   c_DY[NDY]  = {-4, -2, -1, 0, 1, 2, 4};
__constant__ signed char c_DX[33] = {
    -4,-2, 0, 2, 4,          // dy=-4  : o 0..4
    -4,-2, 0, 2, 4,          // dy=-2  : o 5..9
    -1, 0, 1,                // dy=-1  : o 10..12
    -4,-2,-1, 0, 1, 2, 4,    // dy= 0  : o 13..19
    -1, 0, 1,                // dy= 1  : o 20..22
    -4,-2, 0, 2, 4,          // dy= 2  : o 23..27
    -4,-2, 0, 2, 4};         // dy= 4  : o 28..32

__device__ __forceinline__ unsigned short f2bf(float f) {
    unsigned int u = __builtin_bit_cast(unsigned int, f);
    u += 0x7FFFu + ((u >> 16) & 1u);       // round-to-nearest-even
    return (unsigned short)(u >> 16);
}

__global__ __launch_bounds__(TPB)
void dilated_corr_wmma(const float* __restrict__ curr,
                       const float* __restrict__ prev,
                       float* __restrict__ out) {
    extern __shared__ unsigned char smem_raw[];
    unsigned short* curr_lds = (unsigned short*)smem_raw;     // W_*SA ushorts
    unsigned short* prev_lds = curr_lds + W_ * SA;            // NDY*PW*SA ushorts
    float* norm_curr = (float*)(prev_lds + NDY * PW * SA);    // W_ floats
    float* norm_prev = norm_curr + W_;                        // NDY*PW floats
    float* scratch   = norm_prev + NDY * PW;                  // NWAVE*16*SCR floats

    const int tid  = threadIdx.x;
    const int lane = tid & 31;
    const int wave = tid >> 5;
    const int b = blockIdx.x / H_;
    const int y = blockIdx.x % H_;
    const size_t plane = (size_t)H_ * W_;

    // zero the norm accumulators (norm_curr and norm_prev are contiguous)
    for (int i = tid; i < W_ + NDY * PW; i += TPB) norm_curr[i] = 0.f;
    __syncthreads();

    v8f acc[NDY][2];
#pragma unroll
    for (int d = 0; d < NDY; ++d)
#pragma unroll
        for (int t = 0; t < 2; ++t)
#pragma unroll
            for (int e = 0; e < 8; ++e) acc[d][t][e] = 0.f;

    for (int kc = 0; kc < C_; kc += KC) {
        // ---- stage curr row chunk: 384 threads, 2 per x (16 channels each) ----
        {
            int x  = tid >> 1;
            int kh = (tid & 1) * 16;
            const float* gp = curr + (size_t)(b * C_ + kc + kh) * plane
                                   + (size_t)y * W_ + x;
            float ss = 0.f;
#pragma unroll
            for (int j = 0; j < 16; ++j) {
                float v = gp[(size_t)j * plane];
                ss += v * v;
                curr_lds[x * SA + kh + j] = f2bf(v);
            }
            atomicAdd(&norm_curr[x], ss);   // ds_add_f32, conflict-free banks
        }
        // ---- stage 7 prev rows (padded width 200), 32 channels per column ----
        for (int cidx = tid; cidx < NDY * PW; cidx += TPB) {
            int r  = cidx / PW;
            int px = cidx - r * PW;
            int yy = y + c_DY[r];
            int xx = px - 4;
            if (yy >= 0 && yy < H_ && xx >= 0 && xx < W_) {
                const float* gp = prev + (size_t)(b * C_ + kc) * plane
                                       + (size_t)yy * W_ + xx;
                float ss = 0.f;
#pragma unroll
                for (int k = 0; k < KC; ++k) {
                    float v = gp[(size_t)k * plane];
                    ss += v * v;
                    prev_lds[cidx * SA + k] = f2bf(v);
                }
                atomicAdd(&norm_prev[cidx], ss);
            } else {
#pragma unroll
                for (int k = 0; k < KC; ++k) prev_lds[cidx * SA + k] = 0;
            }
        }
        __syncthreads();

        // ---- WMMA: each wave owns x-tile [x0, x0+16) ----
        const int x0 = wave * 16;
        const int M  = lane & 15;
        union V16 { v4u u[2]; v16bf v; };
        V16 A;
        {   // A 16x32 bf16 layout: lane<16 -> K 0..7,16..23 ; lane>=16 -> K 8..15,24..31
            int kb = (lane < 16) ? 0 : 8;
            const unsigned short* ap = &curr_lds[(x0 + M) * SA + kb];
            A.u[0] = *(const v4u*)(ap);
            A.u[1] = *(const v4u*)(ap + 16);
        }
        const int kb2 = (lane < 16) ? 0 : 8;   // B 32x16: lane<16 -> K 0..7,16..23 ; lane>=16 -> K 8..15,24..31
#pragma unroll
        for (int d = 0; d < NDY; ++d) {
#pragma unroll
            for (int t = 0; t < 2; ++t) {
                int px = x0 + t * 16 + M;       // N column = x' + 4 (padded coords)
                V16 Bm;
                const unsigned short* bp = &prev_lds[(d * PW + px) * SA + kb2];
                Bm.u[0] = *(const v4u*)(bp);
                Bm.u[1] = *(const v4u*)(bp + 16);
                acc[d][t] = __builtin_amdgcn_wmma_f32_16x16x32_bf16(
                    false, A.v, false, Bm.v, (short)0, acc[d][t], false, false);
            }
        }
        __syncthreads();   // protect LDS before next chunk's staging
    }

    // ---- extraction: band D[m, m+dx+4], normalize, coalesced store ----
    const int x0   = wave * 16;
    const int col  = lane & 15;
    const int half = lane >> 4;
    float* scr = scratch + wave * 16 * SCR;
    const int OBASE[NDY] = {0, 5, 10, 13, 20, 23, 28};
    const int OCNT[NDY]  = {5, 5, 3, 7, 3, 5, 5};

#pragma unroll
    for (int d = 0; d < NDY; ++d) {
        // spill this dy's two D tiles into per-wave LDS scratch (16 x 32 band)
#pragma unroll
        for (int t = 0; t < 2; ++t)
#pragma unroll
            for (int e = 0; e < 8; ++e)
                scr[(half * 8 + e) * SCR + t * 16 + col] = acc[d][t][e];
        // same-wave LDS ops are in-order; compiler inserts s_wait_dscnt for RAW
        const int cnt = OCNT[d] * 16;
        for (int i = lane; i < cnt; i += 32) {
            int oi = i >> 4, m = i & 15;
            int o  = OBASE[d] + oi;
            int dx = c_DX[o];
            float dot = scr[m * SCR + m + dx + 4];
            float nc  = norm_curr[x0 + m];
            float np  = norm_prev[d * PW + x0 + m + dx + 4];
            float invc = 1.f / fmaxf(sqrtf(nc), 1e-12f);
            float invp = 1.f / fmaxf(sqrtf(np), 1e-12f);
            out[(((size_t)b * 33 + o) * H_ + y) * W_ + x0 + m] = dot * invc * invp;
        }
    }
}

extern "C" void kernel_launch(void* const* d_in, const int* in_sizes, int n_in,
                              void* d_out, int out_size, void* d_ws, size_t ws_size,
                              hipStream_t stream) {
    const float* curr = (const float*)d_in[0];
    const float* prev = (const float*)d_in[1];
    float* out = (float*)d_out;

    const size_t smem = (size_t)(W_ * SA + NDY * PW * SA) * sizeof(unsigned short)
                      + (size_t)(W_ + NDY * PW + NWAVE * 16 * SCR) * sizeof(float);
    // ~155 KB dynamic LDS (<= 160 KB -> 2 workgroups per 320 KB WGP)
    (void)hipFuncSetAttribute((const void*)dilated_corr_wmma,
                              hipFuncAttributeMaxDynamicSharedMemorySize, (int)smem);

    dim3 grid(B_ * H_);
    dim3 block(TPB);
    hipLaunchKernelGGL(dilated_corr_wmma, grid, block, smem, stream, curr, prev, out);
}
